// RNN_28192165331103
// MI455X (gfx1250) — compile-verified, matrix operand order checked by reading, not run on hardware
//
#include <hip/hip_runtime.h>
#include <math.h>

// Problem constants (match reference).
#define Bsz   64
#define Tsz   512
#define Dsz   1024
#define Hsz   1024
#define Gsz   4096      // 4*H
#define LN_EPS 1e-5f

// ---------------------------------------------------------------------------
// CDNA5 WMMA types
// ---------------------------------------------------------------------------
typedef __bf16 v16bf __attribute__((ext_vector_type(16)));
typedef float  f32x8 __attribute__((ext_vector_type(8)));

union TileU { uint4 u[2]; v16bf v; };

// Build a 16x32 bf16 WMMA operand fragment from two contiguous 16B chunks.
// Per ISA layout: lanes 0-15 hold K {0..7, 16..23}, lanes 16-31 hold
// K {8..15, 24..31} for row M = lane&15 (A) / col N = lane&15 (B).
__device__ __forceinline__ v16bf load_frag(const unsigned short* p0,
                                           const unsigned short* p1) {
  TileU t;
  t.u[0] = *reinterpret_cast<const uint4*>(p0);
  t.u[1] = *reinterpret_cast<const uint4*>(p1);
  return t.v;
}

// Same fragment sourced from the XOR-swizzled LDS weight cache.
// s_row points at the 512-short row; key = row & 15; u0 = first 16B unit.
__device__ __forceinline__ v16bf load_frag_lds(const unsigned short* s_row,
                                               int key, int u0) {
  TileU t;
  t.u[0] = *reinterpret_cast<const uint4*>(s_row + (((u0)     ^ key) << 3));
  t.u[1] = *reinterpret_cast<const uint4*>(s_row + (((u0 + 2) ^ key) << 3));
  return t.v;
}

__device__ __forceinline__ f32x8 wmma_bf16(v16bf a, v16bf b, f32x8 c) {
  return __builtin_amdgcn_wmma_f32_16x16x32_bf16(
      false, a, false, b, (short)0, c, false, false);
}

__device__ __forceinline__ unsigned short f2bf(float f) {
  unsigned int u = __float_as_uint(f);
  u += 0x7FFFu + ((u >> 16) & 1u);      // round-to-nearest-even
  return (unsigned short)(u >> 16);
}

__device__ __forceinline__ float sigm(float x) {
  return 1.0f / (1.0f + __expf(-x));
}

// ---------------------------------------------------------------------------
// fp32 -> bf16 conversion
// ---------------------------------------------------------------------------
__global__ void f2bf_kernel(const float* __restrict__ in,
                            unsigned short* __restrict__ out, int n) {
  int i = blockIdx.x * blockDim.x + threadIdx.x;
  if (i < n) out[i] = f2bf(in[i]);
}

__global__ void zero_counter_kernel(unsigned* c) { *c = 0u; }

// ---------------------------------------------------------------------------
// Phase 1: xg[t][b][g] = sum_d x[b][t][d] * w_ih[g][d] + (b_ih[g]+b_hh[g])
// Block = 128 threads = 4 waves. Each wave: 16(M) x 64(N) strip, K=1024.
// Grid: (Gsz/256 N-strips, Tsz*(Bsz/16) M-tiles).
// ---------------------------------------------------------------------------
__global__ __launch_bounds__(128) void xg_gemm_kernel(
    const unsigned short* __restrict__ xb,    // [B,T,D1] bf16
    const unsigned short* __restrict__ wih,   // [4H,D1]  bf16 (B col-major)
    const float* __restrict__ b_ih,
    const float* __restrict__ b_hh,
    float* __restrict__ xg)                   // [T,B,4H] fp32
{
  const int lane  = threadIdx.x & 31;
  const int wave  = threadIdx.x >> 5;
  const int m     = blockIdx.y;               // 0..2047
  const int t     = m >> 2;
  const int btile = (m & 3) << 4;             // 0,16,32,48
  const int nbase = blockIdx.x * 256 + wave * 64;

  const int kh    = (lane >> 4) << 3;         // 0 or 8 (K half select)
  const int colN  = lane & 15;
  const int mrow  = (lane >> 4) << 3;         // C/D row offset per lane half

  const unsigned short* aRow = xb + ((size_t)(btile + colN) * Tsz + t) * Dsz;

  f32x8 acc[4];
  #pragma unroll
  for (int q = 0; q < 4; ++q) acc[q] = (f32x8){0,0,0,0,0,0,0,0};

  for (int k0 = 0; k0 < Dsz; k0 += 32) {
    v16bf a = load_frag(aRow + k0 + kh, aRow + k0 + 16 + kh);
    #pragma unroll
    for (int q = 0; q < 4; ++q) {
      const unsigned short* bRow =
          wih + (size_t)(nbase + q * 16 + colN) * Dsz + k0 + kh;
      v16bf b = load_frag(bRow, bRow + 16);
      acc[q] = wmma_bf16(a, b, acc[q]);
    }
  }

  #pragma unroll
  for (int q = 0; q < 4; ++q) {
    const int n = nbase + q * 16 + colN;
    const float bias = b_ih[n] + b_hh[n];
    #pragma unroll
    for (int i = 0; i < 8; ++i) {
      const int b = btile + mrow + i;         // C row M = i + 8*(lane>>4)
      xg[((size_t)t * Bsz + b) * Gsz + n] = acc[q][i] + bias;
    }
  }
}

// ---------------------------------------------------------------------------
// Phase 2: persistent LSTM scan. 64 WGs x 128 threads (co-resident).
// WG -> 16-column slice j0..j0+15 of H; wave -> 16-row b-block.
// The WG's 64-row w_hh slice is loop-invariant: K[0,512) of it (64 KB) is
// cached once in XOR-swizzled LDS, K[512,1024) streams from L2.
// c stays in registers for all 512 steps; h ping-pongs through L2 in bf16.
// Grid-wide barrier per step via atomic counter + __threadfence().
// ---------------------------------------------------------------------------
__global__ __launch_bounds__(128) void lstm_scan_kernel(
    const unsigned short* __restrict__ whh,   // [4H,H] bf16
    const float* __restrict__ xg,             // [T,B,4H] fp32
    const float* __restrict__ c0,             // [B,H] fp32 (read-only!)
    unsigned short* __restrict__ h_pp,        // [2][B*H] bf16 ping-pong
    float* __restrict__ out,                  // [B,T,H] fp32 (raw h, LN later)
    unsigned* __restrict__ counter)
{
  // 64 rows x 512 bf16 (K first half) = 64 KB, 16B-unit XOR swizzle per row.
  __shared__ unsigned short s_whh[64 * 512];

  const int lane  = threadIdx.x & 31;
  const int wave  = threadIdx.x >> 5;
  const int bbase = wave << 4;
  const int j0    = blockIdx.x << 4;
  const int kh    = (lane >> 4) << 3;
  const int colN  = lane & 15;
  const int mrow  = (lane >> 4) << 3;
  const int j     = j0 + colN;

  // Cooperative fill of the LDS weight cache (once; reused 512 steps).
  for (int idx = threadIdx.x; idx < 64 * 64; idx += 128) {
    const int rl  = idx >> 6;                 // local row 0..63 (gate*16+col)
    const int u   = idx & 63;                 // 16B unit within K[0,512)
    const int q   = rl >> 4;
    const int col = rl & 15;
    const unsigned short* g =
        whh + (size_t)(q * Hsz + j0 + col) * Hsz + u * 8;
    const uint4 d = *reinterpret_cast<const uint4*>(g);
    *reinterpret_cast<uint4*>(&s_whh[rl * 512 + ((u ^ col) << 3)]) = d;
  }

  // B-matrix row pointers for gates i,f,g,o (K second half, from L2).
  const unsigned short* bp[4];
  #pragma unroll
  for (int q = 0; q < 4; ++q)
    bp[q] = whh + (size_t)(q * Hsz + j) * Hsz;

  // Cell state lives in registers: creg[i] <-> C/D element i of this lane.
  float creg[8];
  #pragma unroll
  for (int i = 0; i < 8; ++i)
    creg[i] = c0[(size_t)(bbase + mrow + i) * Hsz + j];

  __syncthreads();

  for (int t = 0; t < Tsz; ++t) {
    const unsigned short* hp = h_pp + (size_t)(t & 1) * (Bsz * Hsz);
    unsigned short*       hn = h_pp + (size_t)((t + 1) & 1) * (Bsz * Hsz);
    const unsigned short* aRow = hp + (size_t)(bbase + colN) * Hsz;

    f32x8 acc0 = (f32x8){0,0,0,0,0,0,0,0};
    f32x8 acc1 = (f32x8){0,0,0,0,0,0,0,0};
    f32x8 acc2 = (f32x8){0,0,0,0,0,0,0,0};
    f32x8 acc3 = (f32x8){0,0,0,0,0,0,0,0};

    // K first half: B fragments from LDS cache (DS pipe), A from L2.
    for (int k0 = 0; k0 < 512; k0 += 32) {
      const int u0 = (k0 >> 3) + (lane >> 4);
      v16bf a = load_frag(aRow + k0 + kh, aRow + k0 + 16 + kh);
      acc0 = wmma_bf16(a, load_frag_lds(&s_whh[(0 * 16 + colN) * 512], colN, u0), acc0);
      acc1 = wmma_bf16(a, load_frag_lds(&s_whh[(1 * 16 + colN) * 512], colN, u0), acc1);
      acc2 = wmma_bf16(a, load_frag_lds(&s_whh[(2 * 16 + colN) * 512], colN, u0), acc2);
      acc3 = wmma_bf16(a, load_frag_lds(&s_whh[(3 * 16 + colN) * 512], colN, u0), acc3);
    }
    // K second half: B fragments stream from L2 (loop-invariant addresses).
    for (int k0 = 512; k0 < Hsz; k0 += 32) {
      v16bf a = load_frag(aRow + k0 + kh, aRow + k0 + 16 + kh);
      acc0 = wmma_bf16(a, load_frag(bp[0] + k0 + kh, bp[0] + k0 + 16 + kh), acc0);
      acc1 = wmma_bf16(a, load_frag(bp[1] + k0 + kh, bp[1] + k0 + 16 + kh), acc1);
      acc2 = wmma_bf16(a, load_frag(bp[2] + k0 + kh, bp[2] + k0 + 16 + kh), acc2);
      acc3 = wmma_bf16(a, load_frag(bp[3] + k0 + kh, bp[3] + k0 + 16 + kh), acc3);
    }

    const float* xgt = xg + (size_t)t * (Bsz * Gsz);
    #pragma unroll
    for (int i = 0; i < 8; ++i) {
      const int b = bbase + mrow + i;
      const size_t gb = (size_t)b * Gsz + j;
      const float gi = sigm (acc0[i] + xgt[gb]);
      const float gf = sigm (acc1[i] + xgt[gb + Hsz]);
      const float gg = tanhf(acc2[i] + xgt[gb + 2 * Hsz]);
      const float go = sigm (acc3[i] + xgt[gb + 3 * Hsz]);
      const float cc = gf * creg[i] + gi * gg;
      creg[i] = cc;
      const float hh = go * tanhf(cc);
      out[((size_t)b * Tsz + t) * Hsz + j] = hh;
      hn[(size_t)b * Hsz + j] = f2bf(hh);
    }

    // Grid-wide barrier: release stores, arrive, spin, acquire.
    __syncthreads();
    if (threadIdx.x == 0) {
      __threadfence();                        // release h writes
      atomicAdd(counter, 1u);
      const unsigned target = (unsigned)(t + 1) * gridDim.x;
      while (atomicAdd(counter, 0u) < target) {
        __builtin_amdgcn_s_sleep(1);
      }
      __threadfence();                        // acquire (inv WGP$)
    }
    __syncthreads();
  }
}

// ---------------------------------------------------------------------------
// Phase 3: in-place LayerNorm over last dim (H=1024). One block per row.
// ---------------------------------------------------------------------------
__global__ __launch_bounds__(256) void ln_kernel(
    float* __restrict__ out, const float* __restrict__ gamma,
    const float* __restrict__ beta)
{
  __shared__ float s1[256];
  __shared__ float s2[256];
  float* p = out + (size_t)blockIdx.x * Hsz;

  float v[4], sum = 0.f, sq = 0.f;
  #pragma unroll
  for (int i = 0; i < 4; ++i) {
    v[i] = p[threadIdx.x + 256 * i];
    sum += v[i];
    sq  += v[i] * v[i];
  }
  s1[threadIdx.x] = sum;
  s2[threadIdx.x] = sq;
  __syncthreads();
  for (int off = 128; off > 0; off >>= 1) {
    if (threadIdx.x < off) {
      s1[threadIdx.x] += s1[threadIdx.x + off];
      s2[threadIdx.x] += s2[threadIdx.x + off];
    }
    __syncthreads();
  }
  const float mean = s1[0] * (1.f / Hsz);
  const float var  = s2[0] * (1.f / Hsz) - mean * mean;
  const float rstd = rsqrtf(var + LN_EPS);
  #pragma unroll
  for (int i = 0; i < 4; ++i) {
    const int jj = threadIdx.x + 256 * i;
    p[jj] = (v[i] - mean) * rstd * gamma[jj] + beta[jj];
  }
}

// ---------------------------------------------------------------------------
// Host launcher (graph-capture safe: only kernel launches on `stream`).
// ---------------------------------------------------------------------------
extern "C" void kernel_launch(void* const* d_in, const int* in_sizes, int n_in,
                              void* d_out, int out_size, void* d_ws,
                              size_t ws_size, hipStream_t stream) {
  (void)in_sizes; (void)n_in; (void)out_size; (void)ws_size;

  const float* x     = (const float*)d_in[0];   // [B,T,D1]
  const float* w_ih  = (const float*)d_in[1];   // [4H,D1]
  const float* w_hh  = (const float*)d_in[2];   // [4H,H]
  const float* b_ih  = (const float*)d_in[3];   // [4H]
  const float* b_hh  = (const float*)d_in[4];   // [4H]
  const float* h0    = (const float*)d_in[5];   // [B,H]
  const float* c0    = (const float*)d_in[6];   // [B,H]
  const float* gamma = (const float*)d_in[7];   // [H]
  const float* beta  = (const float*)d_in[8];   // [H]
  float* out = (float*)d_out;                   // [B,T,H]

  // Workspace carve-up (~621 MB total).
  char* ws = (char*)d_ws;
  unsigned short* x_bf   = (unsigned short*)ws; ws += (size_t)Bsz*Tsz*Dsz*2;
  unsigned short* wih_bf = (unsigned short*)ws; ws += (size_t)Gsz*Dsz*2;
  unsigned short* whh_bf = (unsigned short*)ws; ws += (size_t)Gsz*Hsz*2;
  unsigned short* h_pp   = (unsigned short*)ws; ws += (size_t)2*Bsz*Hsz*2;
  float*          xg     = (float*)ws;          ws += (size_t)Tsz*Bsz*Gsz*4;
  unsigned*       ctr    = (unsigned*)ws;       ws += 128;

  int n;
  n = Bsz * Tsz * Dsz;
  f2bf_kernel<<<(n + 255) / 256, 256, 0, stream>>>(x, x_bf, n);
  n = Gsz * Dsz;
  f2bf_kernel<<<(n + 255) / 256, 256, 0, stream>>>(w_ih, wih_bf, n);
  n = Gsz * Hsz;
  f2bf_kernel<<<(n + 255) / 256, 256, 0, stream>>>(w_hh, whh_bf, n);
  n = Bsz * Hsz;
  f2bf_kernel<<<(n + 255) / 256, 256, 0, stream>>>(h0, h_pp, n);  // buffer 0
  zero_counter_kernel<<<1, 1, 0, stream>>>(ctr);

  // Phase 1: big input-projection GEMM (WMMA bf16, f32 accum).
  xg_gemm_kernel<<<dim3(Gsz / 256, Tsz * (Bsz / 16)), 128, 0, stream>>>(
      x_bf, wih_bf, b_ih, b_hh, xg);

  // Phase 2: persistent 512-step LSTM scan (64 WGs, grid barrier per step).
  lstm_scan_kernel<<<Hsz / 16, 128, 0, stream>>>(
      whh_bf, xg, c0, h_pp, out, ctr);

  // Phase 3: LayerNorm in place on d_out.
  ln_kernel<<<Bsz * Tsz, 256, 0, stream>>>(out, gamma, beta);
}
